// proteusAI_10084583211127
// MI455X (gfx1250) — matrix operand order repaired and checked by reading, not compiled, not run on hardware
//
#include <hip/hip_runtime.h>
#include <hip/hip_bf16.h>
#include <math.h>

// Problem constants (match reference)
#define BB 2
#define NN 2048
#define DD 512
#define HH 8
#define DKK 64

typedef __bf16 bf16_t;
typedef bf16_t v16bf __attribute__((ext_vector_type(16)));
typedef bf16_t v8bf  __attribute__((ext_vector_type(8)));
typedef float  v8f   __attribute__((ext_vector_type(8)));

__device__ __forceinline__ v8f wmma_bf16(v16bf a, v16bf b, v8f c) {
    // D = A(16x32 bf16) * B(32x16 bf16) + C(16x16 f32)
    return __builtin_amdgcn_wmma_f32_16x16x32_bf16(
        /*neg_a=*/false, a, /*neg_b=*/false, b,
        /*c_mod=*/(short)0, c, /*reuse_a=*/false, /*reuse_b=*/false);
}

// Combine two contiguous 8-element bf16 chunks into a 16-element A/B fragment.
__device__ __forceinline__ v16bf cat8(v8bf lo, v8bf hi) {
    return __builtin_shufflevector(lo, hi, 0, 1, 2, 3, 4, 5, 6, 7,
                                           8, 9, 10, 11, 12, 13, 14, 15);
}

__device__ __forceinline__ v8bf cvt8(v8f x) {
    v8bf r;
    #pragma unroll
    for (int i = 0; i < 8; ++i) r[i] = (bf16_t)x[i];   // packs into v_cvt_pk_bf16_f32
    return r;
}

// ---------------------------------------------------------------------------
// Kernel 1: per-head projections. qh = q @ q_proj[h], kh = k @ k_proj[h],
// vhT = (v @ v_proj[h])^T  (DK x N, so attention P*V B-fragments are contiguous)
// Grid: (tiles=(N/16)*(DK/16)=512, B*H=16, 3)   Block: 32 (one wave per tile)
// ---------------------------------------------------------------------------
__global__ __launch_bounds__(32)
void proj_kernel(const float* __restrict__ q, const float* __restrict__ k,
                 const float* __restrict__ v,
                 const float* __restrict__ qp, const float* __restrict__ kp,
                 const float* __restrict__ vp,
                 bf16_t* __restrict__ qh, bf16_t* __restrict__ kh,
                 bf16_t* __restrict__ vhT)
{
    const int lane = threadIdx.x & 31;
    const int half = lane >> 4;
    const int nidx = lane & 15;
    const int which = blockIdx.z;          // 0=q, 1=k, 2=v
    const int bh = blockIdx.y;             // b*H + h
    const int h  = bh % HH;
    const int b  = bh / HH;
    const int tile = blockIdx.x;
    const int mt = tile >> 2;              // 0..127 : 16-row tile of N
    const int nt = tile & 3;               // 0..3   : 16-col tile of DK

    const float* X = (which == 0) ? q : ((which == 1) ? k : v);
    const float* P = (which == 0) ? qp : ((which == 1) ? kp : vp);

    // A-fragment lane row (m = nidx), B-fragment lane column (n = nidx)
    const float* xrow = X + ((size_t)b * NN + (size_t)mt * 16 + nidx) * DD;
    const float* pcol = P + (size_t)h * DD * DKK + (size_t)nt * 16 + nidx;

    v8f acc = {};
    for (int ks = 0; ks < DD / 32; ++ks) {
        const int kb = ks * 32;
        // A: two contiguous 8-float chunks (32B aligned), convert to bf16
        const v8f xa = *(const v8f*)(xrow + kb + 8 * half);        // k = e + 8*half
        const v8f xb = *(const v8f*)(xrow + kb + 16 + 8 * half);   // k = 16 + e + 8*half
        const v16bf a = cat8(cvt8(xa), cvt8(xb));
        // B: stride-DK gather of the projection column (cached in L0/L2)
        v16bf bf;
        #pragma unroll
        for (int e = 0; e < 16; ++e) {
            const int d = kb + ((e < 8) ? 8 * half + e : 16 + 8 * half + (e - 8));
            bf[e] = (bf16_t)pcol[(size_t)d * DKK];
        }
        acc = wmma_bf16(a, bf, acc);
    }

    if (which != 2) {
        bf16_t* dst = ((which == 0) ? qh : kh) + (size_t)bh * NN * DKK;
        #pragma unroll
        for (int j = 0; j < 8; ++j) {
            const int m = mt * 16 + j + 8 * half;                  // row (query index)
            dst[(size_t)m * DKK + nt * 16 + nidx] = (bf16_t)acc[j];
        }
    } else {
        bf16_t* dst = vhT + (size_t)bh * DKK * NN;                 // transposed (DK x N)
        #pragma unroll
        for (int j = 0; j < 8; ++j) {
            const int m  = mt * 16 + j + 8 * half;                 // key index
            const int dk = nt * 16 + nidx;
            dst[(size_t)dk * NN + m] = (bf16_t)acc[j];
        }
    }
}

// ---------------------------------------------------------------------------
// Kernel 2: flash attention with RBF distance modulation.
// One wave per 16-query tile, 4 waves per block. Streams 32 keys / iteration.
// Grid: 512   Block: 128
// ---------------------------------------------------------------------------
__global__ __launch_bounds__(128)
void attn_kernel(const bf16_t* __restrict__ qh, const bf16_t* __restrict__ kh,
                 const bf16_t* __restrict__ vhT,
                 const float* __restrict__ coords,
                 const unsigned char* __restrict__ kpm,
                 bf16_t* __restrict__ ctx)
{
    // per-wave P staging; 36-float row pad: 16B-aligned rows, 16 distinct banks
    __shared__ float plds[4][16][36];

    const int lane = threadIdx.x & 31;
    const int wid  = threadIdx.x >> 5;
    const int half = lane >> 4;
    const int nidx = lane & 15;

    const int tileId = blockIdx.x * 4 + wid;       // 0..2047
    const int qt = tileId & (NN / 16 - 1);
    const int bh = tileId / (NN / 16);
    const int b  = bh / HH;
    const int h  = bh % HH;
    const int q0 = qt * 16;

    // per-head spread: MIN + (BASE^t - 1)/(BASE-1) * (MAX-MIN), t = h*(1-eps)/(H-1)
    const float t = 0.98f * (float)h / 7.0f;
    const float spread = 3.7f + (powf(20.0f, t) - 1.0f) * (16.3f / 19.0f);
    const float s3 = 3.0f * spread;
    const float inv2s2 = 1.0f / (2.0f * spread * spread);
    const float NEG_INF = -__builtin_inff();

    const bf16_t* qbase = qh  + (size_t)bh * NN * DKK;
    const bf16_t* kbase = kh  + (size_t)bh * NN * DKK;
    const bf16_t* vbase = vhT + (size_t)bh * DKK * NN;

    // Q A-fragments for DK=64 (two K=32 slabs), lane row m = nidx
    v16bf a0, a1;
    {
        const bf16_t* qrow = qbase + (size_t)(q0 + nidx) * DKK;
        a0 = cat8(*(const v8bf*)(qrow + 8 * half),
                  *(const v8bf*)(qrow + 16 + 8 * half));
        a1 = cat8(*(const v8bf*)(qrow + 32 + 8 * half),
                  *(const v8bf*)(qrow + 48 + 8 * half));
    }

    // query coords for the 8 rows this lane's C-fragment elements cover
    float qcx[8], qcy[8], qcz[8];
    #pragma unroll
    for (int j = 0; j < 8; ++j) {
        const int m = q0 + j + 8 * half;
        const float* cp = coords + ((size_t)b * NN + m) * 3;
        qcx[j] = cp[0]; qcy[j] = cp[1]; qcz[j] = cp[2];
    }

    float rmax[8], rsum[8];
    #pragma unroll
    for (int j = 0; j < 8; ++j) { rmax[j] = NEG_INF; rsum[j] = 0.0f; }
    v8f o0 = {}, o1 = {}, o2 = {}, o3 = {};

    for (int kb = 0; kb < NN; kb += 32) {
        // prefetch next key/value tiles while this block computes
        if (kb + 32 < NN) {
            __builtin_prefetch(kbase + (size_t)(kb + 32 + nidx) * DKK, 0, 1);
            __builtin_prefetch(kbase + (size_t)(kb + 48 + nidx) * DKK, 0, 1);
            __builtin_prefetch(vbase + (size_t)nidx * NN + kb + 32, 0, 1);
        }

        float l0[8], l1[8];
        #pragma unroll
        for (int sub = 0; sub < 2; ++sub) {
            const int key = kb + sub * 16 + nidx;          // B-fragment lane column
            const bf16_t* krow = kbase + (size_t)key * DKK;
            const v16bf b0 = cat8(*(const v8bf*)(krow + 8 * half), *(const v8bf*)(krow + 16 + 8 * half));
            const v16bf b1 = cat8(*(const v8bf*)(krow + 32 + 8 * half), *(const v8bf*)(krow + 48 + 8 * half));
            v8f s = {};
            s = wmma_bf16(a0, b0, s);
            s = wmma_bf16(a1, b1, s);

            const float* cp = coords + ((size_t)b * NN + key) * 3;
            const float kx = cp[0], ky = cp[1], kz = cp[2];
            const bool kp = kpm[(size_t)b * NN + key] != 0;
            float* lo = sub ? l1 : l0;
            #pragma unroll
            for (int j = 0; j < 8; ++j) {
                float av = s[j] * 0.125f;                  // / sqrt(DK)
                const float dx = qcx[j] - kx, dy = qcy[j] - ky, dz = qcz[j] - kz;
                const float dist = sqrtf(dx * dx + dy * dy + dz * dz);
                const float d1 = dist < spread ? spread : dist;
                const bool dmask = d1 > s3;
                const float d2 = dmask ? s3 : d1;
                float rbf = __expf(-(d2 * d2) * inv2s2);
                rbf = (av < 0.0f) ? 1.0f / rbf : rbf;
                av *= rbf;
                lo[j] = (kp || dmask) ? NEG_INF : av;
            }
        }

        // online softmax update (row reductions across the 16-lane N group)
        float nmax[8], corr[8];
        #pragma unroll
        for (int j = 0; j < 8; ++j) {
            float tm = fmaxf(l0[j], l1[j]);
            #pragma unroll
            for (int mk = 1; mk < 16; mk <<= 1) tm = fmaxf(tm, __shfl_xor(tm, mk, 32));
            nmax[j] = fmaxf(rmax[j], tm);
            corr[j] = (rmax[j] == NEG_INF) ? 0.0f : __expf(rmax[j] - nmax[j]);
        }

        #pragma unroll
        for (int j = 0; j < 8; ++j) {
            const float p0 = (l0[j] == NEG_INF) ? 0.0f : __expf(l0[j] - nmax[j]);
            const float p1 = (l1[j] == NEG_INF) ? 0.0f : __expf(l1[j] - nmax[j]);
            plds[wid][j + 8 * half][nidx]      = p0;       // C layout: row j+8h, col nidx
            plds[wid][j + 8 * half][16 + nidx] = p1;
            float ps = p0 + p1;
            #pragma unroll
            for (int mk = 1; mk < 16; mk <<= 1) ps += __shfl_xor(ps, mk, 32);
            rsum[j] = rsum[j] * corr[j] + ps;
            rmax[j] = nmax[j];
        }
        #pragma unroll
        for (int j = 0; j < 8; ++j) {
            o0[j] *= corr[j]; o1[j] *= corr[j]; o2[j] *= corr[j]; o3[j] *= corr[j];
        }

        __syncthreads();                                   // P visible in LDS

        // re-gather P in A-fragment layout (lane row m = nidx), vector ds loads
        v16bf pa;
        {
            const float* prow = &plds[wid][nidx][0];       // 16B-aligned row
            const float4 c0 = *(const float4*)(prow + 8 * half);
            const float4 c1 = *(const float4*)(prow + 8 * half + 4);
            const float4 c2 = *(const float4*)(prow + 16 + 8 * half);
            const float4 c3 = *(const float4*)(prow + 16 + 8 * half + 4);
            pa[0]  = (bf16_t)c0.x; pa[1]  = (bf16_t)c0.y;
            pa[2]  = (bf16_t)c0.z; pa[3]  = (bf16_t)c0.w;
            pa[4]  = (bf16_t)c1.x; pa[5]  = (bf16_t)c1.y;
            pa[6]  = (bf16_t)c1.z; pa[7]  = (bf16_t)c1.w;
            pa[8]  = (bf16_t)c2.x; pa[9]  = (bf16_t)c2.y;
            pa[10] = (bf16_t)c2.z; pa[11] = (bf16_t)c2.w;
            pa[12] = (bf16_t)c3.x; pa[13] = (bf16_t)c3.y;
            pa[14] = (bf16_t)c3.z; pa[15] = (bf16_t)c3.w;
        }

        // O += P(16x32) @ V(32x64), four 16-col output chunks (contiguous b128 loads)
        #pragma unroll
        for (int c = 0; c < 4; ++c) {
            const bf16_t* vrow = vbase + (size_t)(c * 16 + nidx) * NN + kb;
            const v16bf bv = cat8(*(const v8bf*)(vrow + 8 * half), *(const v8bf*)(vrow + 16 + 8 * half));
            if      (c == 0) o0 = wmma_bf16(pa, bv, o0);
            else if (c == 1) o1 = wmma_bf16(pa, bv, o1);
            else if (c == 2) o2 = wmma_bf16(pa, bv, o2);
            else             o3 = wmma_bf16(pa, bv, o3);
        }
        __syncthreads();                                   // before next overwrite
    }

    // normalize and store context (bf16, heads concatenated -> (B,N,D))
    #pragma unroll
    for (int j = 0; j < 8; ++j) {
        const float inv = 1.0f / rsum[j];
        const size_t row = (size_t)b * NN + q0 + j + 8 * half;
        bf16_t* dst = ctx + row * DD + (size_t)h * DKK;
        dst[ 0 + nidx] = (bf16_t)(o0[j] * inv);
        dst[16 + nidx] = (bf16_t)(o1[j] * inv);
        dst[32 + nidx] = (bf16_t)(o2[j] * inv);
        dst[48 + nidx] = (bf16_t)(o3[j] * inv);
    }
}

// ---------------------------------------------------------------------------
// Kernel 3: out = ctx @ out_w^T + out_b.  M=B*N=4096, N=512, K=512.
// Grid: (256, 32)  Block: 32 (one wave per 16x16 tile)
// ---------------------------------------------------------------------------
__global__ __launch_bounds__(32)
void outproj_kernel(const bf16_t* __restrict__ ctx, const float* __restrict__ w,
                    const float* __restrict__ bias, float* __restrict__ out)
{
    const int lane = threadIdx.x & 31;
    const int half = lane >> 4;
    const int nidx = lane & 15;
    const int mt = blockIdx.x;             // 0..255
    const int nt = blockIdx.y;             // 0..31

    const bf16_t* arow = ctx + (size_t)(mt * 16 + nidx) * DD;
    const float*  wrow = w   + (size_t)(nt * 16 + nidx) * DD;   // B[k][n] = w[n_glob][k]

    v8f acc = {};
    for (int ks = 0; ks < DD / 32; ++ks) {
        const int kb = ks * 32;
        const v16bf a = cat8(*(const v8bf*)(arow + kb + 8 * half),
                             *(const v8bf*)(arow + kb + 16 + 8 * half));
        const v8f w0 = *(const v8f*)(wrow + kb + 8 * half);
        const v8f w1 = *(const v8f*)(wrow + kb + 16 + 8 * half);
        const v16bf bf = cat8(cvt8(w0), cvt8(w1));
        acc = wmma_bf16(a, bf, acc);
    }

    const float bb = bias[nt * 16 + nidx];
    #pragma unroll
    for (int j = 0; j < 8; ++j) {
        const size_t row = (size_t)mt * 16 + j + 8 * half;
        out[row * DD + nt * 16 + nidx] = acc[j] + bb;
    }
}

// ---------------------------------------------------------------------------
extern "C" void kernel_launch(void* const* d_in, const int* in_sizes, int n_in,
                              void* d_out, int out_size, void* d_ws, size_t ws_size,
                              hipStream_t stream) {
    (void)in_sizes; (void)n_in; (void)out_size; (void)ws_size;

    const float* q      = (const float*)d_in[0];
    const float* k      = (const float*)d_in[1];
    const float* v      = (const float*)d_in[2];
    const float* coords = (const float*)d_in[3];
    const unsigned char* kpm = (const unsigned char*)d_in[4];   // bool mask, 1B/elem
    const float* qp     = (const float*)d_in[5];
    const float* kp     = (const float*)d_in[6];
    const float* vp     = (const float*)d_in[7];
    const float* out_w  = (const float*)d_in[8];
    const float* out_b  = (const float*)d_in[9];
    float* out = (float*)d_out;

    // workspace carve-up (bf16): qh | kh | vhT | ctx  (4 MB each, 16 MB total)
    const size_t HEAD_ELEMS = (size_t)BB * HH * NN * DKK;       // 2,097,152
    bf16_t* qh  = (bf16_t*)d_ws;
    bf16_t* kh  = qh  + HEAD_ELEMS;
    bf16_t* vhT = kh  + HEAD_ELEMS;
    bf16_t* ctx = vhT + HEAD_ELEMS;                             // B*N*D elems

    // 1) projections: (N/16)*(DK/16)=512 tiles x (B*H)=16 x 3 matrices
    proj_kernel<<<dim3(512, BB * HH, 3), 32, 0, stream>>>(q, k, v, qp, kp, vp,
                                                          qh, kh, vhT);
    // 2) flash attention w/ RBF: 2048 query tiles, 4 waves per block
    attn_kernel<<<dim3(512), 128, 0, stream>>>(qh, kh, vhT, coords, kpm, ctx);
    // 3) output projection + bias
    outproj_kernel<<<dim3(256, 32), 32, 0, stream>>>(ctx, out_w, out_b, out);
}
